// Network_68547678044943
// MI455X (gfx1250) — matrix operand order repaired and checked by reading, not compiled, not run on hardware
//
#include <hip/hip_runtime.h>

typedef __attribute__((ext_vector_type(16))) _Float16 v16h;
typedef __attribute__((ext_vector_type(8)))  _Float16 v8h;
typedef __attribute__((ext_vector_type(8)))  float    v8f;

#define NPTS 262144
#define CCH  48
#define DD   32
#define KK   4
#define G0   512
#define G1   512
#define G2   128
#define KT   192   /* KK * CCH stacked reduction dim */

// workspace byte offsets (all 16B aligned)
#define LT0_OFF 0                      // [4][512][48] f16 = 196608 B
#define LT1_OFF 196608                 // [4][512][48] f16 = 196608 B
#define LT2_OFF 393216                 // [4][128][48] f16 =  49152 B
#define BT_OFF  442368                 // [3][32][192] f16 =  36864 B
// total ws use: 479232 B

// ---------------------------------------------------------------------------
// Prep: gather topk pages, transpose lines to [k][g][c] (f16), and build
// per-plane weight matrices transposed [d][j] (f16) with topk_weight folded.
// ---------------------------------------------------------------------------
__global__ void pose_prep_kernel(const int*   __restrict__ topk_id,
                                 const float* __restrict__ topk_w,
                                 const float* __restrict__ line0,
                                 const float* __restrict__ line1,
                                 const float* __restrict__ line2,
                                 const float* __restrict__ feat0,
                                 const float* __restrict__ feat1,
                                 const float* __restrict__ feat2,
                                 _Float16* __restrict__ lt0,
                                 _Float16* __restrict__ lt1,
                                 _Float16* __restrict__ lt2,
                                 _Float16* __restrict__ bt)
{
  const int idx = blockIdx.x * blockDim.x + threadIdx.x;
  const int E0 = KK * G0 * CCH;            // 98304
  const int E1 = KK * G1 * CCH;            // 98304
  const int E2 = KK * G2 * CCH;            // 24576
  const int EB = 3 * DD * KT;              // 18432

  if (idx < E0) {
    int k = idx / (G0 * CCH); int rem = idx % (G0 * CCH);
    int g = rem / CCH;        int c   = rem % CCH;
    lt0[idx] = (_Float16)line0[(topk_id[k] * CCH + c) * G0 + g];
  } else if (idx < E0 + E1) {
    int l = idx - E0;
    int k = l / (G1 * CCH);   int rem = l % (G1 * CCH);
    int g = rem / CCH;        int c   = rem % CCH;
    lt1[l] = (_Float16)line1[(topk_id[k] * CCH + c) * G1 + g];
  } else if (idx < E0 + E1 + E2) {
    int l = idx - E0 - E1;
    int k = l / (G2 * CCH);   int rem = l % (G2 * CCH);
    int g = rem / CCH;        int c   = rem % CCH;
    lt2[l] = (_Float16)line2[(topk_id[k] * CCH + c) * G2 + g];
  } else if (idx < E0 + E1 + E2 + EB) {
    int l = idx - E0 - E1 - E2;
    int p = l / (DD * KT);    int rem  = l % (DD * KT);
    int dcol = rem / KT;      int j    = rem % KT;
    int k = j / CCH;          int c    = j % CCH;
    const float* f = (p == 0) ? feat0 : (p == 1) ? feat1 : feat2;
    bt[l] = (_Float16)(topk_w[k] * f[(topk_id[k] * CCH + c) * DD + dcol]);
  }
}

// ---------------------------------------------------------------------------
// Main: each wave computes a 16-point tile: out[16][96] via 36 WMMAs
// (6 K-chunks x 3 planes x 2 N-tiles). Lerp + coordinate products run in
// packed f16 (v_pk_fma_f16 / v_pk_mul_f16) -- no f32<->f16 conversions.
// ---------------------------------------------------------------------------
__global__ void __launch_bounds__(256)
pose_main_kernel(const float*    __restrict__ pts,
                 const _Float16* __restrict__ lt0,
                 const _Float16* __restrict__ lt1,
                 const _Float16* __restrict__ lt2,
                 const _Float16* __restrict__ btg,
                 float*          __restrict__ out)
{
  __shared__ __align__(32) _Float16 sB[3 * DD * KT];   // 36864 B

  // cooperative stage of the 3 transposed weight matrices into LDS
  {
    const uint4* src = (const uint4*)btg;
    uint4*       dst = (uint4*)sB;
    const int n16 = (3 * DD * KT * 2) / 16;            // 2304 x 16B
    for (int i = threadIdx.x; i < n16; i += blockDim.x) dst[i] = src[i];
  }
  __syncthreads();

  const int  lane   = threadIdx.x & 31;
  const int  wid    = threadIdx.x >> 5;
  const bool hi     = lane >= 16;
  const int  colrow = lane & 15;
  const int  tile   = blockIdx.x * 8 + wid;            // 16 points per tile
  const int  npt    = tile * 16 + colrow;              // this lane's point

  // per-point interpolation parameters (index math in f32, weight -> f16)
  float px[3];
  px[0] = pts[npt * 3 + 0];
  px[1] = pts[npt * 3 + 1];
  px[2] = pts[npt * 3 + 2];
  const int Gd[3] = {G0, G1, G2};
  int i0v[3]; v8h whv[3];
#pragma unroll
  for (int d = 0; d < 3; ++d) {
    float pos = (px[d] + 1.0f) * 0.5f * (float)(Gd[d] - 1);
    int i0 = (int)floorf(pos);
    i0 = i0 < 0 ? 0 : (i0 > Gd[d] - 2 ? Gd[d] - 2 : i0);
    i0v[d] = i0;
    _Float16 w = (_Float16)(pos - (float)i0);
    v8h ws = {w, w, w, w, w, w, w, w};
    whv[d] = ws;
  }
  const _Float16* lts[3] = {lt0, lt1, lt2};

  const v8f vzero = {0.f, 0.f, 0.f, 0.f, 0.f, 0.f, 0.f, 0.f};
  v8f acc[3][2];
#pragma unroll
  for (int p = 0; p < 3; ++p) {
#pragma unroll
    for (int t = 0; t < 2; ++t) acc[p][t] = vzero;
  }

#pragma unroll
  for (int c6 = 0; c6 < 6; ++c6) {
    // Build the three A fragments (one per plane) for this 32-wide K chunk.
    // f16 A 16x32 layout: lanes<16 hold K runs {0..7,16..23}; lanes>=16 hold
    // {8..15,24..31}. Runs of 8 are aligned and never cross an expert (48=6x8).
    v8h a0[2], a1[2], a2[2];
#pragma unroll
    for (int r = 0; r < 2; ++r) {
      const int jrun = 32 * c6 + r * 16 + (hi ? 8 : 0);
      const int k  = jrun / CCH;
      const int cc = jrun - k * CCH;
      v8h cd[3];
#pragma unroll
      for (int d = 0; d < 3; ++d) {
        const _Float16* bp = lts[d] + (k * Gd[d] + i0v[d]) * CCH + cc;
        v8h h0 = *(const v8h*)bp;
        v8h h1 = *(const v8h*)(bp + CCH);
        cd[d] = h0 + whv[d] * (h1 - h0);     // packed f16 lerp
      }
      a0[r] = cd[0] * cd[1];                 // packed f16 products
      a1[r] = cd[0] * cd[2];
      a2[r] = cd[1] * cd[2];
    }
    const v16h A0 = __builtin_shufflevector(a0[0], a0[1],
        0, 1, 2, 3, 4, 5, 6, 7, 8, 9, 10, 11, 12, 13, 14, 15);
    const v16h A1 = __builtin_shufflevector(a1[0], a1[1],
        0, 1, 2, 3, 4, 5, 6, 7, 8, 9, 10, 11, 12, 13, 14, 15);
    const v16h A2 = __builtin_shufflevector(a2[0], a2[1],
        0, 1, 2, 3, 4, 5, 6, 7, 8, 9, 10, 11, 12, 13, 14, 15);

    // B fragments from LDS: f16 B 32x16 layout = lane holds column N=lane%16,
    // K runs (lane<16 ? {0..7,16..23} : {8..15,24..31}).
    const int koff = 32 * c6 + (hi ? 8 : 0);
#pragma unroll
    for (int p = 0; p < 3; ++p) {
      const v16h Ap = (p == 0) ? A0 : (p == 1) ? A1 : A2;
#pragma unroll
      for (int t = 0; t < 2; ++t) {
        const _Float16* bq = sB + ((p * DD + t * 16 + colrow) * KT + koff);
        const v16h Bpt = __builtin_shufflevector(*(const v8h*)bq, *(const v8h*)(bq + 16),
            0, 1, 2, 3, 4, 5, 6, 7, 8, 9, 10, 11, 12, 13, 14, 15);
        acc[p][t] = __builtin_amdgcn_wmma_f32_16x16x32_f16(
            false, Ap, false, Bpt, (short)0, acc[p][t], false, false);
      }
    }
  }

  // Store: C/D layout — element v of lane l is (M = v + (l>=16 ? 8 : 0),
  // Ncol = l%16).  col = p*32 + t*16 + Ncol,  n = tile*16 + M.
  // Non-temporal: output is streamed, never re-read; keep L2 for the tables.
  const int Mb = hi ? 8 : 0;
#pragma unroll
  for (int p = 0; p < 3; ++p) {
#pragma unroll
    for (int t = 0; t < 2; ++t) {
      const int col = p * DD + t * 16 + colrow;
#pragma unroll
      for (int v = 0; v < 8; ++v) {
        const int n = tile * 16 + Mb + v;
        __builtin_nontemporal_store(acc[p][t][v], &out[n * 96 + col]);
      }
    }
  }
}

// ---------------------------------------------------------------------------
extern "C" void kernel_launch(void* const* d_in, const int* in_sizes, int n_in,
                              void* d_out, int out_size, void* d_ws, size_t ws_size,
                              hipStream_t stream) {
  const float* pts     = (const float*)d_in[0];
  const int*   topk_id = (const int*)  d_in[1];
  const float* topk_w  = (const float*)d_in[2];
  const float* line0   = (const float*)d_in[3];
  const float* line1   = (const float*)d_in[4];
  const float* line2   = (const float*)d_in[5];
  const float* feat0   = (const float*)d_in[6];
  const float* feat1   = (const float*)d_in[7];
  const float* feat2   = (const float*)d_in[8];
  float* outp = (float*)d_out;

  char* ws = (char*)d_ws;
  _Float16* lt0 = (_Float16*)(ws + LT0_OFF);
  _Float16* lt1 = (_Float16*)(ws + LT1_OFF);
  _Float16* lt2 = (_Float16*)(ws + LT2_OFF);
  _Float16* bt  = (_Float16*)(ws + BT_OFF);

  // prep: 98304+98304+24576+18432 = 239616 elements -> 936 blocks of 256
  pose_prep_kernel<<<936, 256, 0, stream>>>(topk_id, topk_w,
                                            line0, line1, line2,
                                            feat0, feat1, feat2,
                                            lt0, lt1, lt2, bt);

  // main: 262144 points / (8 waves * 16 pts) = 2048 blocks of 256
  pose_main_kernel<<<2048, 256, 0, stream>>>(pts, lt0, lt1, lt2, bt, outp);
}
